// SingleStageModel1_61134564491868
// MI455X (gfx1250) — compile-verified, matrix operand order checked by reading, not run on hardware
//
#include <hip/hip_runtime.h>
#include <hip/hip_bf16.h>
#include <math.h>

typedef __attribute__((ext_vector_type(16))) __bf16 v16bf;
typedef __attribute__((ext_vector_type(8)))  float  v8f;

#define Bc    32
#define Tc    256
#define DIMc  2048
#define Dc    128
#define INNERc 256
#define NHm   4
#define DHm   64
#define NHs   4
#define DHs   32
#define FFc   192
#define NCLSc 19
#define Mrows (Bc*Tc)   // 8192

// ---------------------------------------------------------------------------
// WMMA 16-bit fragment helpers (ISA 7.12.2, wave32).
// For element i (0..15), half = lane>>4 the K index is:
//   k0 + half*8 + (i&7) + (i>=8 ? 16 : 0)
// i.e. two contiguous 8-float runs at p[0..7] and p[16..23] where
// p = base + k0 + half*8.  Load them as 4x float4 (b128).
// ---------------------------------------------------------------------------
template <typename PT>
__device__ __forceinline__ v16bf frag_load(PT p) {
  const float4 x0 = *(const float4*)(p);
  const float4 x1 = *(const float4*)(p + 4);
  const float4 x2 = *(const float4*)(p + 16);
  const float4 x3 = *(const float4*)(p + 20);
  v16bf f;
  f[0]  = (__bf16)x0.x; f[1]  = (__bf16)x0.y; f[2]  = (__bf16)x0.z; f[3]  = (__bf16)x0.w;
  f[4]  = (__bf16)x1.x; f[5]  = (__bf16)x1.y; f[6]  = (__bf16)x1.z; f[7]  = (__bf16)x1.w;
  f[8]  = (__bf16)x2.x; f[9]  = (__bf16)x2.y; f[10] = (__bf16)x2.z; f[11] = (__bf16)x2.w;
  f[12] = (__bf16)x3.x; f[13] = (__bf16)x3.y; f[14] = (__bf16)x3.z; f[15] = (__bf16)x3.w;
  return f;
}

// masked variant (mask = 0.0f zeroes the fragment; avoids divergent loads)
__device__ __forceinline__ v16bf frag_load_m(const float* __restrict__ p, float m) {
  const float4 x0 = *(const float4*)(p);
  const float4 x1 = *(const float4*)(p + 4);
  const float4 x2 = *(const float4*)(p + 16);
  const float4 x3 = *(const float4*)(p + 20);
  v16bf f;
  f[0]  = (__bf16)(x0.x * m); f[1]  = (__bf16)(x0.y * m); f[2]  = (__bf16)(x0.z * m); f[3]  = (__bf16)(x0.w * m);
  f[4]  = (__bf16)(x1.x * m); f[5]  = (__bf16)(x1.y * m); f[6]  = (__bf16)(x1.z * m); f[7]  = (__bf16)(x1.w * m);
  f[8]  = (__bf16)(x2.x * m); f[9]  = (__bf16)(x2.y * m); f[10] = (__bf16)(x2.z * m); f[11] = (__bf16)(x2.w * m);
  f[12] = (__bf16)(x3.x * m); f[13] = (__bf16)(x3.y * m); f[14] = (__bf16)(x3.z * m); f[15] = (__bf16)(x3.w * m);
  return f;
}

// ---------------------------------------------------------------------------
// Generic GEMM: C[m,n] = sum_k A[m,k]*W[n,k] (+bias[n]) (+resid[m,n])
// CONTIG:  A[m*a_rs + k]                (vectorized b128 loads)
// !CONTIG: A[(m/a_rpb)*a_bs + (m%a_rpb)*a_rs + k*a_ks]
// out_mode 0: C[m*N+n] ; out_mode 1: C[b*N*T + n*T + t] with m = b*T+t
// One 16x16 tile per wave, 4 waves per block. K multiple of 32.
// ---------------------------------------------------------------------------
template <bool CONTIG>
__global__ __launch_bounds__(128)
void k_gemm(const float* __restrict__ A, const float* __restrict__ W,
            const float* __restrict__ bias, const float* __restrict__ resid,
            float* __restrict__ C, int M, int N, int K,
            int a_rpb, long long a_bs, long long a_rs, long long a_ks,
            int out_mode, int T_out)
{
  int wave = threadIdx.x >> 5, lane = threadIdx.x & 31;
  int ntn = (N + 15) >> 4;
  long long tile = (long long)blockIdx.x * 4 + wave;
  long long total = (long long)((M + 15) >> 4) * ntn;
  if (tile >= total) return;
  int tm = (int)(tile / ntn), tn = (int)(tile % ntn);
  int m0 = tm * 16, n0 = tn * 16;
  int half = lane >> 4, l16 = lane & 15;
  int mrow = m0 + l16;
  int nrow = n0 + l16;
  float wmask = (nrow < N) ? 1.0f : 0.0f;
  const float* wrow = W + (long long)((nrow < N) ? nrow : 0) * K + half * 8;
  long long abase;
  if (CONTIG) abase = (long long)mrow * a_rs;
  else abase = (long long)(mrow / a_rpb) * a_bs + (long long)(mrow % a_rpb) * a_rs;

  v8f acc = {};
  for (int k0 = 0; k0 < K; k0 += 32) {
    __builtin_prefetch(wrow + k0 + 32, 0, 1);
    v16bf af, bf;
    if (CONTIG) {
      af = frag_load(A + abase + k0 + half * 8);
    } else {
#pragma unroll
      for (int i = 0; i < 16; ++i) {
        int kk = k0 + half * 8 + (i & 7) + ((i >= 8) ? 16 : 0);
        af[i] = (__bf16)A[abase + (long long)kk * a_ks];
      }
    }
    bf = frag_load_m(wrow + k0, wmask);
    acc = __builtin_amdgcn_wmma_f32_16x16x32_bf16(false, af, false, bf,
                                                  (short)0, acc, false, false);
  }
  int n = n0 + l16;
#pragma unroll
  for (int r = 0; r < 8; ++r) {
    int m = m0 + r + 8 * half;
    if (n < N && m < M) {
      float val = acc[r];
      if (bias)  val += bias[n];
      if (resid) val += resid[(long long)m * N + n];
      long long o;
      if (out_mode == 0) o = (long long)m * N + n;
      else { int b = m / T_out, t = m % T_out; o = ((long long)b * N + n) * T_out + t; }
      C[o] = val;
    }
  }
}

// ---------------------------------------------------------------------------
// LayerNorm / group-norm over last dim Dd (32/64/128); weight row selected by
// ((row/wdiv)%wmod)*Dd. One wave per row.
// ---------------------------------------------------------------------------
__global__ __launch_bounds__(256)
void k_norm(const float* __restrict__ X, const float* __restrict__ w,
            float* __restrict__ Y, long long R, int Dd, int wdiv, int wmod)
{
  int wv = threadIdx.x >> 5, lane = threadIdx.x & 31;
  long long row = (long long)blockIdx.x * 8 + wv;
  if (row >= R) return;
  const float* x = X + row * Dd;
  int per = Dd >> 5;
  float v[4]; float s = 0.f, s2 = 0.f;
  for (int i = 0; i < per; ++i) { float t = x[lane + (i << 5)]; v[i] = t; s += t; s2 += t * t; }
  for (int m = 16; m >= 1; m >>= 1) { s += __shfl_xor(s, m, 32); s2 += __shfl_xor(s2, m, 32); }
  float mu = s / Dd, var = s2 / Dd - mu * mu;
  float inv = rsqrtf(var + 1e-5f);
  const float* wp = w + (long long)((row / wdiv) % wmod) * Dd;
  float* y = Y + row * Dd;
  for (int i = 0; i < per; ++i) { int c = lane + (i << 5); y[c] = (v[i] - mu) * inv * wp[c]; }
}

// depthwise causal conv (K=4) + SiLU; X rows have stride in_rs, Y contiguous [B,S,C]
__global__ __launch_bounds__(256)
void k_conv_silu(const float* __restrict__ X, const float* __restrict__ w,
                 const float* __restrict__ bias, float* __restrict__ Y,
                 int Bb, int S, int C, long long in_rs)
{
  long long idx = (long long)blockIdx.x * 256 + threadIdx.x;
  if (idx >= (long long)Bb * S * C) return;
  int c = (int)(idx % C); long long bs = idx / C;
  int t = (int)(bs % S);  int b = (int)(bs / S);
  const float* xb = X + (long long)b * S * in_rs + c;
  float acc = bias[c];
#pragma unroll
  for (int k = 0; k < 4; ++k) { int tt = t - 3 + k; if (tt >= 0) acc += xb[(long long)tt * in_rs] * w[c * 4 + k]; }
  Y[((long long)b * S + t) * C + c] = acc / (1.f + __expf(-acc));
}

// block-diagonal linear (LinearHeadwiseExpand)
__global__ __launch_bounds__(256)
void k_headwise(const float* __restrict__ X, const float* __restrict__ W,
                float* __restrict__ Y, long long rows, int H, int dout, int din,
                long long in_rs)
{
  long long idx = (long long)blockIdx.x * 256 + threadIdx.x;
  long long total = rows * (long long)H * dout;
  if (idx >= total) return;
  int o = (int)(idx % dout); long long t = idx / dout;
  int hh = (int)(t % H);     long long row = t / H;
  const float* x  = X + row * in_rs + (long long)hh * din;
  const float* wp = W + ((long long)hh * dout + o) * din;
  float acc = 0.f;
  for (int i = 0; i < din; ++i) acc += x[i] * wp[i];
  Y[row * (long long)(H * dout) + hh * dout + o] = acc;
}

// ig/fg gate projections from (q,k,v) concat; one wave per (b,s,h)
__global__ __launch_bounds__(256)
void k_igfg(const float* __restrict__ q, const float* __restrict__ k,
            const float* __restrict__ v,
            const float* __restrict__ igw, const float* __restrict__ igb,
            const float* __restrict__ fgw, const float* __restrict__ fgb,
            float* __restrict__ ig, float* __restrict__ fg,
            int Bb, int S, int NH, int IN)
{
  int wv = threadIdx.x >> 5, lane = threadIdx.x & 31;
  long long wid = (long long)blockIdx.x * 8 + wv;
  if (wid >= (long long)Bb * S * NH) return;
  int hh = (int)(wid % NH); long long row = wid / NH;
  const float* iw = igw + (long long)hh * 3 * IN;
  const float* fw = fgw + (long long)hh * 3 * IN;
  float si = 0.f, sf = 0.f;
  for (int i = lane; i < IN; i += 32) {
    float qv = q[row * IN + i], kv = k[row * IN + i], vv = v[row * IN + i];
    si += qv * iw[i] + kv * iw[IN + i] + vv * iw[2 * IN + i];
    sf += qv * fw[i] + kv * fw[IN + i] + vv * fw[2 * IN + i];
  }
  for (int m = 16; m >= 1; m >>= 1) { si += __shfl_xor(si, m, 32); sf += __shfl_xor(sf, m, 32); }
  if (lane == 0) {
    int b = (int)(row / S), s = (int)(row % S);
    long long o = ((long long)b * NH + hh) * S + s;
    ig[o] = si + igb[hh];
    fg[o] = sf + fgb[hh];
  }
}

// per-(b,h) decay scan: a[s]=ig[s]-cumsum(logsig(fg)); runmax; maxD=c+runmax
__global__ void k_decay(const float* __restrict__ ig, const float* __restrict__ fg,
                        float* __restrict__ a, float* __restrict__ rm,
                        float* __restrict__ maxD, int n, int S)
{
  int i = blockIdx.x * blockDim.x + threadIdx.x;
  if (i >= n) return;
  const float* igp = ig + (long long)i * S;
  const float* fgp = fg + (long long)i * S;
  float c = 0.f, run = -1e30f;
  for (int s = 0; s < S; ++s) {
    float f = fgp[s];
    float ls = fminf(f, 0.f) - log1pf(__expf(-fabsf(f)));
    c += ls;
    float av = igp[s] - c;
    run = fmaxf(run, av);
    a[(long long)i * S + s]    = av;
    rm[(long long)i * S + s]   = run;
    maxD[(long long)i * S + s] = c + run;
  }
}

// V transpose for the S*V WMMA: Vt[b,h,d,s] = V[b,s,h*DH+d]
__global__ __launch_bounds__(256)
void k_vt(const float* __restrict__ V, float* __restrict__ Vt,
          int Bb, int S, int NH, int DH)
{
  long long idx = (long long)blockIdx.x * 256 + threadIdx.x;
  if (idx >= (long long)Bb * S * NH * DH) return;
  int d = (int)(idx % DH); long long t = idx / DH;
  int hh = (int)(t % NH);  long long bs = t / NH;
  int s = (int)(bs % S);   int b = (int)(bs / S);
  Vt[(((long long)b * NH + hh) * DH + d) * S + s] = V[idx];
}

// ---------------------------------------------------------------------------
// mLSTM parallel attention, WMMA. 2 waves/block, each wave owns one
// (b,h,16-row q-tile). Scores staged in LDS between QK^T and S*V WMMAs.
// Vt is [B,NH,DH,S] so both GEMMs use vectorized fragment loads.
// ---------------------------------------------------------------------------
__global__ __launch_bounds__(64)
void k_attn(const float* __restrict__ Q, const float* __restrict__ Km,
            const float* __restrict__ Vt,
            const float* __restrict__ a, const float* __restrict__ rm,
            const float* __restrict__ maxD, float* __restrict__ O,
            int Bb, int NH, int S, int DH)
{
  __shared__ float sc[2][16 * Tc];
  int wv = threadIdx.x >> 5, lane = threadIdx.x & 31;
  int tile = blockIdx.x * 2 + wv;
  int nqt = S >> 4;
  int qt = tile % nqt, bh = tile / nqt;
  int hh = bh % NH, b = bh / NH;
  int half = lane >> 4, l16 = lane & 15;
  int m0 = qt * 16;
  int IN = NH * DH;
  const float* qrow = Q + ((long long)b * S + (m0 + l16)) * IN + hh * DH + half * 8;
  v16bf aq0 = frag_load(qrow);
  v16bf aq1 = frag_load(qrow + 32);

  const float* av  = a  + (long long)bh * S;
  const float* rmv = rm + (long long)bh * S;
  float myrm[8], rs[8];
#pragma unroll
  for (int r = 0; r < 8; ++r) { myrm[r] = rmv[m0 + r + 8 * half]; rs[r] = 0.f; }
  float* scw = sc[wv];
  const float scale = rsqrtf((float)DH);

  for (int kt = 0; kt < nqt; ++kt) {
    int s0 = kt * 16;
    const float* krow = Km + ((long long)b * S + (s0 + l16)) * IN + hh * DH + half * 8;
    v16bf bk0 = frag_load(krow);
    v16bf bk1 = frag_load(krow + 32);
    v8f acc = {};
    acc = __builtin_amdgcn_wmma_f32_16x16x32_bf16(false, aq0, false, bk0, (short)0, acc, false, false);
    acc = __builtin_amdgcn_wmma_f32_16x16x32_bf16(false, aq1, false, bk1, (short)0, acc, false, false);
    int s = s0 + l16;
    float as = av[s];
#pragma unroll
    for (int r = 0; r < 8; ++r) {
      int m = m0 + r + 8 * half;
      // branchless causal mask: divergent lanes make a guarded exp pay for
      // both paths anyway; v_cndmask beats 8 execz branches per key tile.
      float e = __expf(as - myrm[r]);
      float val = (s <= m) ? acc[r] * scale * e : 0.f;
      rs[r] += val;
      scw[(r + 8 * half) * Tc + s] = val;
    }
  }
  // row-sum reduce across the 16 lanes of each half
#pragma unroll
  for (int r = 0; r < 8; ++r) {
    float t = rs[r];
    t += __shfl_xor(t, 1, 32); t += __shfl_xor(t, 2, 32);
    t += __shfl_xor(t, 4, 32); t += __shfl_xor(t, 8, 32);
    rs[r] = t;
  }
  const float* mdv = maxD + (long long)bh * S;
  float inv[8];
#pragma unroll
  for (int r = 0; r < 8; ++r) {
    int m = m0 + r + 8 * half;
    float nrmv = fmaxf(fabsf(rs[r]), __expf(-mdv[m]));
    inv[r] = 1.f / (nrmv + 1e-6f);
  }
#pragma unroll
  for (int r = 0; r < 8; ++r) {
    int ml = r + 8 * half;
    for (int s = l16; s < S; s += 16) scw[ml * Tc + s] *= inv[r];
  }
  // out = scores @ V  (A from LDS, B from Vt; both b128 fragment loads)
  const float* vtb = Vt + (long long)bh * DH * S;
  v8f zv = {};
  v8f oacc[4] = { zv, zv, zv, zv };
  for (int k0 = 0; k0 < S; k0 += 32) {
    v16bf af = frag_load(scw + l16 * Tc + k0 + half * 8);
#pragma unroll
    for (int j = 0; j < 4; ++j) {
      v16bf bf = frag_load(vtb + (long long)(j * 16 + l16) * S + k0 + half * 8);
      oacc[j] = __builtin_amdgcn_wmma_f32_16x16x32_bf16(false, af, false, bf, (short)0, oacc[j], false, false);
    }
  }
#pragma unroll
  for (int j = 0; j < 4; ++j) {
    int d = j * 16 + l16;
#pragma unroll
    for (int r = 0; r < 8; ++r) {
      int m = m0 + r + 8 * half;
      O[(((long long)b * NH + hh) * S + m) * DH + d] = oacc[j][r];
    }
  }
}

// mLSTM epilogue: hs = (hc_norm + skip*xa) * silu(z); hc in [B,NH,S,DH]
__global__ __launch_bounds__(256)
void k_mix(const float* __restrict__ hc, const float* __restrict__ xa,
           const float* __restrict__ xi, const float* __restrict__ skip,
           float* __restrict__ out, int Bb, int S, int NH, int DH)
{
  int IN = NH * DH;
  long long idx = (long long)blockIdx.x * 256 + threadIdx.x;
  if (idx >= (long long)Bb * S * IN) return;
  int c = (int)(idx % IN); long long bs = idx / IN;
  int s = (int)(bs % S), b = (int)(bs / S);
  int hh = c / DH, d = c % DH;
  float hv = hc[(((long long)b * NH + hh) * S + s) * DH + d] + skip[c] * xa[idx];
  float z = xi[bs * (long long)(2 * IN) + IN + c];
  out[idx] = hv * (z / (1.f + __expf(-z)));
}

// sLSTM recurrent scan; one block per batch, thread = (head,o) state element.
__global__ __launch_bounds__(128)
void k_slstm(const float* __restrict__ gi, const float* __restrict__ gf,
             const float* __restrict__ gz, const float* __restrict__ go,
             const float* __restrict__ rec, const float* __restrict__ bias,
             float* __restrict__ hs, int S)
{
  int b = blockIdx.x, tid = threadIdx.x;
  int hh = tid >> 5;
  float rw[4][32];
#pragma unroll
  for (int g = 0; g < 4; ++g)
    for (int d = 0; d < 32; ++d)
      rw[g][d] = rec[(((long long)g * 4 + hh) * 32 + (tid & 31)) * 32 + d];
  float bsv[4];
#pragma unroll
  for (int g = 0; g < 4; ++g) bsv[g] = bias[((long long)g * 4 + hh) * 32 + (tid & 31)];
  __shared__ float hp[128];
  hp[tid] = 0.f;
  __syncthreads();
  float cst = 0.f, nst = 0.f, mst = 0.f;
  for (int s = 0; s < S; ++s) {
    float r0 = bsv[0], r1 = bsv[1], r2 = bsv[2], r3 = bsv[3];
    const float* hph = hp + hh * 32;
    for (int d = 0; d < 32; ++d) {
      float hv = hph[d];
      r0 += hv * rw[0][d]; r1 += hv * rw[1][d];
      r2 += hv * rw[2][d]; r3 += hv * rw[3][d];
    }
    long long base = ((long long)b * S + s) * 128 + tid;
    float iraw = gi[base] + r0, fraw = gf[base] + r1;
    float zraw = gz[base] + r2, oraw = go[base] + r3;
    float lfm = mst + (fminf(fraw, 0.f) - log1pf(__expf(-fabsf(fraw))));
    float mnew = fmaxf(iraw, lfm);
    float igt = __expf(iraw - mnew), fgt = __expf(lfm - mnew);
    cst = fgt * cst + igt * tanhf(zraw);
    nst = fgt * nst + igt;
    float hnew = (1.f / (1.f + __expf(-oraw))) * cst / nst;
    mst = mnew;
    __syncthreads();
    hp[tid] = hnew;
    hs[base] = hnew;
    __syncthreads();
  }
}

__global__ __launch_bounds__(256)
void k_add(float* __restrict__ X, const float* __restrict__ Y, long long n)
{
  long long i = (long long)blockIdx.x * 256 + threadIdx.x;
  if (i < n) X[i] += Y[i];
}

// gated FFN middle: out = gelu_exact(gate) * up
__global__ __launch_bounds__(256)
void k_gelugate(const float* __restrict__ U, float* __restrict__ Y,
                long long rows, int F)
{
  long long idx = (long long)blockIdx.x * 256 + threadIdx.x;
  if (idx >= rows * F) return;
  int j = (int)(idx % F); long long r = idx / F;
  float g = U[r * (long long)(2 * F) + j];
  float u = U[r * (long long)(2 * F) + F + j];
  float ge = 0.5f * g * (1.f + erff(g * 0.70710678118654752f));
  Y[idx] = ge * u;
}

// feats[b,d,t] = X[(b*T+t)*D + d]
__global__ __launch_bounds__(256)
void k_feats(const float* __restrict__ X, float* __restrict__ Y, int Bb, int T, int Dd)
{
  long long idx = (long long)blockIdx.x * 256 + threadIdx.x;
  if (idx >= (long long)Bb * T * Dd) return;
  int d = (int)(idx % Dd); long long bt = idx / Dd;
  int t = (int)(bt % T), b = (int)(bt / T);
  Y[((long long)b * Dd + d) * T + t] = X[idx];
}

// ---------------------------------------------------------------------------
extern "C" void kernel_launch(void* const* d_in, const int* in_sizes, int n_in,
                              void* d_out, int out_size, void* d_ws, size_t ws_size,
                              hipStream_t stream)
{
  const float* x        = (const float*)d_in[0];
  const float* w_in     = (const float*)d_in[1];
  const float* b_in     = (const float*)d_in[2];
  const float* m_ln_w   = (const float*)d_in[3];
  const float* m_up_w   = (const float*)d_in[4];
  const float* m_conv_w = (const float*)d_in[5];
  const float* m_conv_b = (const float*)d_in[6];
  const float* m_q_w    = (const float*)d_in[7];
  const float* m_k_w    = (const float*)d_in[8];
  const float* m_v_w    = (const float*)d_in[9];
  const float* m_ig_w   = (const float*)d_in[10];
  const float* m_ig_b   = (const float*)d_in[11];
  const float* m_fg_w   = (const float*)d_in[12];
  const float* m_fg_b   = (const float*)d_in[13];
  const float* m_skip   = (const float*)d_in[14];
  const float* m_norm_w = (const float*)d_in[15];
  const float* m_down_w = (const float*)d_in[16];
  const float* s_ln_w   = (const float*)d_in[17];
  const float* s_conv_w = (const float*)d_in[18];
  const float* s_conv_b = (const float*)d_in[19];
  const float* s_gate_w = (const float*)d_in[20];
  const float* s_rec_w  = (const float*)d_in[21];
  const float* s_bias   = (const float*)d_in[22];
  const float* s_gn_w   = (const float*)d_in[23];
  const float* f_ln_w   = (const float*)d_in[24];
  const float* f_up_w   = (const float*)d_in[25];
  const float* f_down_w = (const float*)d_in[26];
  const float* post_ln  = (const float*)d_in[27];
  const float* w_out    = (const float*)d_in[28];
  const float* b_out    = (const float*)d_in[29];
  (void)in_sizes; (void)n_in; (void)out_size; (void)ws_size;

  float* ws  = (float*)d_ws;
  float* h   = ws;                 // [8192,128]
  float* tln = h   + 1048576;      // [8192,128]
  float* A1  = tln + 1048576;      // [8192,512]
  float* A2  = A1  + 4194304;      // [8192,256]
  float* A3  = A2  + 2097152;
  float* A4  = A3  + 2097152;
  float* A5  = A4  + 2097152;
  float* A6  = A5  + 2097152;
  float* A7  = A6  + 2097152;      // Vt [B,NH,DH,S]
  float* Gig = A7  + 2097152;      // [B*NH*S] each
  float* Gfg = Gig + 32768;
  float* Ga  = Gfg + 32768;
  float* Grm = Ga  + 32768;
  float* Gmd = Grm + 32768;

  auto gemm_blocks = [](int M, int N) { return ((((M + 15) / 16) * ((N + 15) / 16)) + 3) / 4; };

  // ---- input conv_1x1: h[b,t,f] = sum_d x[b,d,t] * w_in[f,d] + b_in[f]
  k_gemm<false><<<gemm_blocks(Mrows, Dc), 128, 0, stream>>>(
      x, w_in, b_in, nullptr, h, Mrows, Dc, DIMc,
      Tc, (long long)DIMc * Tc, 1LL, (long long)Tc, 0, 0);

  auto run_mlstm = [&](int mi) {
    k_norm<<<Mrows / 8, 256, 0, stream>>>(h, m_ln_w + mi * Dc, tln, Mrows, Dc, 1, 1);
    k_gemm<true><<<gemm_blocks(Mrows, 2 * INNERc), 128, 0, stream>>>(
        tln, m_up_w + (long long)mi * 2 * INNERc * Dc, nullptr, nullptr, A1,
        Mrows, 2 * INNERc, Dc, Mrows, 0LL, (long long)Dc, 1LL, 0, 0);
    k_conv_silu<<<(Mrows * INNERc + 255) / 256, 256, 0, stream>>>(
        A1, m_conv_w + mi * INNERc * 4, m_conv_b + mi * INNERc, A2,
        Bc, Tc, INNERc, (long long)(2 * INNERc));
    long long hwT = (long long)Mrows * INNERc;
    k_headwise<<<(int)((hwT + 255) / 256), 256, 0, stream>>>(
        A2, m_q_w + mi * 64 * 16, A3, (long long)Mrows, 64, 4, 4, (long long)INNERc);
    k_headwise<<<(int)((hwT + 255) / 256), 256, 0, stream>>>(
        A2, m_k_w + mi * 64 * 16, A4, (long long)Mrows, 64, 4, 4, (long long)INNERc);
    k_headwise<<<(int)((hwT + 255) / 256), 256, 0, stream>>>(
        A1, m_v_w + mi * 64 * 16, A5, (long long)Mrows, 64, 4, 4, (long long)(2 * INNERc));
    k_igfg<<<(Bc * Tc * NHm) / 8, 256, 0, stream>>>(
        A3, A4, A5, m_ig_w + mi * NHm * 3 * INNERc, m_ig_b + mi * NHm,
        m_fg_w + mi * NHm * 3 * INNERc, m_fg_b + mi * NHm,
        Gig, Gfg, Bc, Tc, NHm, INNERc);
    k_decay<<<1, 128, 0, stream>>>(Gig, Gfg, Ga, Grm, Gmd, Bc * NHm, Tc);
    k_vt<<<(Mrows * INNERc + 255) / 256, 256, 0, stream>>>(A5, A7, Bc, Tc, NHm, DHm);
    k_attn<<<(Bc * NHm * (Tc / 16)) / 2, 64, 0, stream>>>(
        A3, A4, A7, Ga, Grm, Gmd, A6, Bc, NHm, Tc, DHm);
    k_norm<<<(Bc * NHm * Tc) / 8, 256, 0, stream>>>(
        A6, m_norm_w + mi * INNERc, A6, (long long)Bc * NHm * Tc, DHm, Tc, NHm);
    k_mix<<<(Mrows * INNERc + 255) / 256, 256, 0, stream>>>(
        A6, A2, A1, m_skip + mi * INNERc, A2, Bc, Tc, NHm, DHm);
    k_gemm<true><<<gemm_blocks(Mrows, Dc), 128, 0, stream>>>(
        A2, m_down_w + (long long)mi * Dc * INNERc, nullptr, h, h,
        Mrows, Dc, INNERc, Mrows, 0LL, (long long)INNERc, 1LL, 0, 0);
  };

  auto run_slstm = [&]() {
    k_norm<<<Mrows / 8, 256, 0, stream>>>(h, s_ln_w, tln, Mrows, Dc, 1, 1);
    k_conv_silu<<<(Mrows * Dc + 255) / 256, 256, 0, stream>>>(
        tln, s_conv_w, s_conv_b, A2, Bc, Tc, Dc, (long long)Dc);
    long long tot = (long long)Mrows * Dc;
    k_headwise<<<(int)((tot + 255) / 256), 256, 0, stream>>>(
        A2, s_gate_w + 0 * 4096, A3, (long long)Mrows, NHs, DHs, DHs, (long long)Dc);
    k_headwise<<<(int)((tot + 255) / 256), 256, 0, stream>>>(
        A2, s_gate_w + 1 * 4096, A4, (long long)Mrows, NHs, DHs, DHs, (long long)Dc);
    k_headwise<<<(int)((tot + 255) / 256), 256, 0, stream>>>(
        tln, s_gate_w + 2 * 4096, A5, (long long)Mrows, NHs, DHs, DHs, (long long)Dc);
    k_headwise<<<(int)((tot + 255) / 256), 256, 0, stream>>>(
        tln, s_gate_w + 3 * 4096, A6, (long long)Mrows, NHs, DHs, DHs, (long long)Dc);
    k_slstm<<<Bc, 128, 0, stream>>>(A3, A4, A5, A6, s_rec_w, s_bias, A2, Tc);
    k_norm<<<(Bc * Tc * NHs) / 8, 256, 0, stream>>>(
        A2, s_gn_w, A2, (long long)Bc * Tc * NHs, DHs, 1, NHs);
    k_add<<<(1048576 + 255) / 256, 256, 0, stream>>>(h, A2, 1048576LL);
    k_norm<<<Mrows / 8, 256, 0, stream>>>(h, f_ln_w, tln, Mrows, Dc, 1, 1);
    k_gemm<true><<<gemm_blocks(Mrows, 2 * FFc), 128, 0, stream>>>(
        tln, f_up_w, nullptr, nullptr, A1, Mrows, 2 * FFc, Dc,
        Mrows, 0LL, (long long)Dc, 1LL, 0, 0);
    k_gelugate<<<(int)(((long long)Mrows * FFc + 255) / 256), 256, 0, stream>>>(
        A1, A5, (long long)Mrows, FFc);
    k_gemm<true><<<gemm_blocks(Mrows, Dc), 128, 0, stream>>>(
        A5, f_down_w, nullptr, h, h, Mrows, Dc, FFc,
        Mrows, 0LL, (long long)FFc, 1LL, 0, 0);
  };

  // block order: mLSTM, sLSTM, then 5 more mLSTM
  run_mlstm(0);
  run_slstm();
  for (int mi = 1; mi < 6; ++mi) run_mlstm(mi);

  // final LN + heads
  k_norm<<<Mrows / 8, 256, 0, stream>>>(h, post_ln, tln, Mrows, Dc, 1, 1);
  float* dout = (float*)d_out;
  k_gemm<true><<<gemm_blocks(Mrows, NCLSc), 128, 0, stream>>>(
      tln, w_out, b_out, nullptr, dout, Mrows, NCLSc, Dc,
      Mrows, 0LL, (long long)Dc, 1LL, 1, Tc);
  k_feats<<<(1048576 + 255) / 256, 256, 0, stream>>>(
      tln, dout + (long long)Bc * NCLSc * Tc, Bc, Tc, Dc);
}
